// HeteroGraphConv_76141180223897
// MI455X (gfx1250) — matrix operand order repaired and checked by reading, not compiled, not run on hardware
//
#include <hip/hip_runtime.h>
#include <hip/hip_bf16.h>
#include <math.h>

// ---------------------------------------------------------------------------
// Types for WMMA (CDNA5 / gfx1250, wave32)
// ---------------------------------------------------------------------------
typedef __attribute__((ext_vector_type(16))) __bf16 v16bf;
typedef __attribute__((ext_vector_type(8)))  __bf16 v8bf;
typedef __attribute__((ext_vector_type(8)))  float  v8f;

#define DIN 256
#define HF  512
#define NHEAD 8
#define FOUT 64
#define NEG_SLOPE 0.2f

// Use CDNA5 async global->LDS copies (ASYNCcnt) for GEMM A-tile staging.
#if defined(__gfx1250__)
#define USE_ASYNC_LDS 1
#else
#define USE_ASYNC_LDS 0
#endif

// ---------------------------------------------------------------------------
// Helpers
// ---------------------------------------------------------------------------
__device__ __forceinline__ unsigned short f32_to_bf16_rne(float f) {
  unsigned u = __float_as_uint(f);
  u += 0x7FFFu + ((u >> 16) & 1u);
  return (unsigned short)(u >> 16);
}
// order-preserving map float -> uint (for atomicMax on float values)
__device__ __forceinline__ unsigned f2ord(float f) {
  unsigned u = __float_as_uint(f);
  return (u & 0x80000000u) ? ~u : (u | 0x80000000u);
}
__device__ __forceinline__ float ord2f(unsigned u) {
  unsigned v = (u & 0x80000000u) ? (u & 0x7FFFFFFFu) : ~u;
  return __uint_as_float(v);
}
__device__ __forceinline__ float lrelu(float v) {
  return v > 0.0f ? v : NEG_SLOPE * v;
}

// ---------------------------------------------------------------------------
// Small prep kernels
// ---------------------------------------------------------------------------
__global__ void cvt_bf16_kernel(const float* __restrict__ src,
                                unsigned short* __restrict__ dst, int n) {
  int t = blockIdx.x * blockDim.x + threadIdx.x;
  if (t < n) dst[t] = f32_to_bf16_rne(src[t]);
}

// WvT[r][n][k] = bf16(Wv[r][k][n]);  Wv: [3][256][512], WvT: [3][512][256]
__global__ void transpose_w_kernel(const float* __restrict__ Wv,
                                   unsigned short* __restrict__ WvT) {
  int t = blockIdx.x * blockDim.x + threadIdx.x;  // 3*512*256
  if (t >= 3 * HF * DIN) return;
  int r = t / (HF * DIN);
  int rem = t - r * (HF * DIN);
  int n = rem / DIN;
  int k = rem - n * DIN;
  WvT[t] = f32_to_bf16_rne(Wv[(size_t)r * DIN * HF + (size_t)k * HF + n]);
}

// Wp[r][k][h] = sum_j Wv[r][k][j] * Wqk[r][j][h]   (fused Wv@Wq / Wv@Wk)
__global__ void fuse_w_kernel(const float* __restrict__ Wv,
                              const float* __restrict__ Wqk,
                              float* __restrict__ Wp) {
  int t = blockIdx.x * blockDim.x + threadIdx.x;  // 3*256*8
  if (t >= 3 * DIN * NHEAD) return;
  int r = t / (DIN * NHEAD);
  int rem = t - r * (DIN * NHEAD);
  int k = rem / NHEAD;
  int h = rem - k * NHEAD;
  const float* wv = Wv + (size_t)r * DIN * HF + (size_t)k * HF;
  const float* wq = Wqk + (size_t)r * HF * NHEAD + h;
  float s = 0.f;
  for (int j = 0; j < HF; ++j) s += wv[j] * wq[(size_t)j * NHEAD];
  Wp[t] = s;
}

// bp[r][h] = b2[r][h] + sum_j bv[r][j]*Wqk[r][j][h]
__global__ void fuse_b_kernel(const float* __restrict__ bv,
                              const float* __restrict__ Wqk,
                              const float* __restrict__ b2,
                              float* __restrict__ bp) {
  int t = blockIdx.x * blockDim.x + threadIdx.x;  // 24
  if (t >= 3 * NHEAD) return;
  int r = t / NHEAD, h = t - r * NHEAD;
  const float* bvr = bv + (size_t)r * HF;
  const float* wq = Wqk + (size_t)r * HF * NHEAD + h;
  float s = b2[t];
  for (int j = 0; j < HF; ++j) s += bvr[j] * wq[(size_t)j * NHEAD];
  bp[t] = s;
}

__global__ void init_f32_kernel(float* p, float v, int n) {
  int t = blockIdx.x * blockDim.x + threadIdx.x;
  if (t < n) p[t] = v;
}
__global__ void init_u32_kernel(unsigned* p, unsigned v, int n) {
  int t = blockIdx.x * blockDim.x + threadIdx.x;
  if (t < n) p[t] = v;
}

// ---------------------------------------------------------------------------
// Big GEMM: C[M,512] = A[M,256](bf16) @ B[256,512](bf16, pre-transposed as
// BT[512,256]) + bias[512]; f32 accumulate via V_WMMA_F32_16X16X32_BF16.
//
// Block = 256 threads = 8 waves; one block covers 16 rows x ALL 512 cols:
// wave w owns n-tiles 4w..4w+3 and reuses one A fragment across 4 WMMAs.
// The 16x32 A tile is double-buffered in LDS and staged with
// GLOBAL_LOAD_ASYNC_TO_LDS_B32 (ASYNCcnt + s_wait_asynccnt), one barrier
// per K-step: the top-of-step barrier both publishes the async tile and
// proves all waves consumed the other buffer (each wave's ds reads drain
// via DScnt before its WMMAs issue, which precede its barrier arrival).
// ---------------------------------------------------------------------------
__global__ __launch_bounds__(256) void gemm_bf16_wmma_kernel(
    const unsigned short* __restrict__ A,   // [M,256] bf16
    const unsigned short* __restrict__ BT,  // [512,256] bf16 (B^T)
    const float* __restrict__ bias,         // [512]
    float* __restrict__ C, int M) {
  __shared__ __align__(32) unsigned short As[2][16 * 32];

  const int tid = threadIdx.x;
  const int lane = tid & 31;
  const int wid = tid >> 5;
  const int m0 = blockIdx.x * 16;
  const int nloc = lane & 15;

  const unsigned short* brow[4];
  v8f acc[4];
#pragma unroll
  for (int j = 0; j < 4; ++j) {
    int ng = wid * 64 + j * 16 + nloc;
    brow[j] = BT + (size_t)ng * DIN;
    float bval = bias[ng];
#pragma unroll
    for (int i = 0; i < 8; ++i) acc[j][i] = bval;  // bias depends only on col
  }

  // staging coords: each thread moves one dword (2 bf16) of the 16x32 tile
  const int arow_ld = tid >> 4;        // 0..15
  const int acol_ld = (tid & 15) * 2;  // 0,2,..,30
  const unsigned short* ag = A + (size_t)(m0 + arow_ld) * DIN + acol_ld;

  const int kbl = (lane < 16) ? 0 : 16;  // A lane K-half (contiguous, matches B)
  const int kbh = (lane < 16) ? 0 : 16;  // B lane K-half (contiguous, matches A)

#define STAGE_A(KK, P)                                                       \
  do {                                                                       \
    if (USE_ASYNC_LDS) {                                                     \
      unsigned ldsa =                                                        \
          (unsigned)(uintptr_t)&As[(P)][arow_ld * 32 + acol_ld];             \
      unsigned long long ga = (unsigned long long)(uintptr_t)(ag + (KK));    \
      asm volatile("global_load_async_to_lds_b32 %0, %1, off" ::"v"(ldsa),   \
                   "v"(ga)                                                   \
                   : "memory");                                              \
    } else {                                                                 \
      *(unsigned*)&As[(P)][arow_ld * 32 + acol_ld] =                         \
          *(const unsigned*)(ag + (KK));                                     \
    }                                                                        \
  } while (0)

  STAGE_A(0, 0);
  int p = 0;
  for (int kk = 0; kk < DIN; kk += 32) {
#if USE_ASYNC_LDS
    asm volatile("s_wait_asynccnt 0x0" ::: "memory");
#endif
    __syncthreads();  // tile for this step visible; other buffer free
    if (kk + 32 < DIN) STAGE_A(kk + 32, p ^ 1);

    // A fragment: elems 0..15 = K kb..kb+15 (contiguous)
    v8bf lo = *(const v8bf*)&As[p][nloc * 32 + kbl];
    v8bf hi = *(const v8bf*)&As[p][nloc * 32 + kbl + 8];
    v16bf a;
#pragma unroll
    for (int i = 0; i < 8; ++i) { a[i] = lo[i]; a[i + 8] = hi[i]; }

#pragma unroll
    for (int j = 0; j < 4; ++j) {
      v16bf b = *(const v16bf*)(brow[j] + kk + kbh);  // 16 contiguous K
      acc[j] = __builtin_amdgcn_wmma_f32_16x16x32_bf16(
          false, a, false, b, (short)0, acc[j], false, false);
    }
    p ^= 1;
  }
#undef STAGE_A

  // C layout: VGPR i -> row m0+i (lanes 0-15) / m0+8+i (lanes 16-31)
  const int mb = m0 + ((lane < 16) ? 0 : 8);
#pragma unroll
  for (int j = 0; j < 4; ++j) {
    int ng = wid * 64 + j * 16 + nloc;
#pragma unroll
    for (int i = 0; i < 8; ++i) C[(size_t)(mb + i) * HF + ng] = acc[j][i];
  }
}

// ---------------------------------------------------------------------------
// Skinny fused GEMM: out[n,h] = bp[h] + sum_k x[n,k] * Wp[k,h]  ([N,256]@[256,8])
// ---------------------------------------------------------------------------
__global__ void qk_gemm_kernel(const float* __restrict__ x,
                               const float* __restrict__ Wp,
                               const float* __restrict__ bp,
                               float* __restrict__ out, int N) {
  int t = blockIdx.x * blockDim.x + threadIdx.x;
  if (t >= N * NHEAD) return;
  int n = t >> 3, h = t & 7;
  const float* xr = x + (size_t)n * DIN;
  float s = bp[h];
  for (int k = 0; k < DIN; ++k) s += xr[k] * Wp[k * NHEAD + h];
  out[t] = s;
}

// ---------------------------------------------------------------------------
// Edge passes (thread per (edge, head) for max/sum; wave per edge for scatter)
// ---------------------------------------------------------------------------
__global__ void edge_max_kernel(const int* __restrict__ src,
                                const int* __restrict__ dst,
                                const float* __restrict__ q,
                                const float* __restrict__ k,
                                unsigned* __restrict__ mord, int E) {
  int t = blockIdx.x * blockDim.x + threadIdx.x;
  if (t >= E * NHEAD) return;
  int e = t >> 3, h = t & 7;
  int d = dst[e], s = src[e];
  float v = lrelu(q[d * NHEAD + h] + k[s * NHEAD + h]);
  atomicMax(&mord[d * NHEAD + h], f2ord(v));
}

// mf = finite(ord2f(mord)) ? v : 0 ; denom = 0
__global__ void m_final_kernel(const unsigned* __restrict__ mord,
                               float* __restrict__ mf,
                               float* __restrict__ denom, int n) {
  int t = blockIdx.x * blockDim.x + threadIdx.x;
  if (t >= n) return;
  float v = ord2f(mord[t]);
  bool fin = (v == v) && (fabsf(v) <= 3.0e38f);
  mf[t] = fin ? v : 0.0f;
  denom[t] = 0.0f;
}

__global__ void edge_sum_kernel(const int* __restrict__ src,
                                const int* __restrict__ dst,
                                const float* __restrict__ q,
                                const float* __restrict__ k,
                                const float* __restrict__ mf,
                                float* __restrict__ denom, int E) {
  int t = blockIdx.x * blockDim.x + threadIdx.x;
  if (t >= E * NHEAD) return;
  int e = t >> 3, h = t & 7;
  int d = dst[e], s = src[e];
  float v = lrelu(q[d * NHEAD + h] + k[s * NHEAD + h]);
  atomicAdd(&denom[d * NHEAD + h], __expf(v - mf[d * NHEAD + h]));
}

// wave32 per edge: lanes 0-7 compute per-head attn; all 32 lanes stream the
// 512-float source row with coalesced loads + f32 global atomics into agg.
__global__ __launch_bounds__(256) void edge_scatter_kernel(
    const int* __restrict__ src, const int* __restrict__ dst,
    const float* __restrict__ q, const float* __restrict__ k,
    const float* __restrict__ mf, const float* __restrict__ denom,
    const float* __restrict__ hsrc, float* __restrict__ agg, int E) {
  int wid = threadIdx.x >> 5;
  int lane = threadIdx.x & 31;
  int e = blockIdx.x * 8 + wid;
  if (e >= E) return;
  int d = dst[e], s = src[e];

  const float* hs = hsrc + (size_t)s * HF;
  __builtin_prefetch(hs + lane, 0, 1);          // global_prefetch of src row
  __builtin_prefetch(hs + 256 + lane, 0, 1);

  float attn = 0.0f;
  if (lane < NHEAD) {
    int i = d * NHEAD + lane;
    float v = lrelu(q[i] + k[s * NHEAD + lane]);
    attn = __expf(v - mf[i]) / fmaxf(denom[i], 1e-9f);
  }

  float* ag = agg + (size_t)d * HF;
#pragma unroll
  for (int i = 0; i < 16; ++i) {          // 16 chunks of 32 -> 512 = H*F
    float a = __shfl(attn, i >> 1, 32);   // head = chunk/2 (64 floats/head)
    int idx = i * 32 + lane;
    atomicAdd(&ag[idx], a * hs[idx]);
  }
}

// out[n,f] = scale * sum_h agg[n,h,f]
__global__ void head_mean_kernel(const float* __restrict__ agg,
                                 float* __restrict__ out, float scale, int N) {
  int t = blockIdx.x * blockDim.x + threadIdx.x;
  if (t >= N * FOUT) return;
  int n = t >> 6, f = t & 63;
  const float* a = agg + (size_t)n * HF + f;
  float s = 0.f;
#pragma unroll
  for (int h = 0; h < NHEAD; ++h) s += a[h * FOUT];
  out[t] = s * scale;
}

// ---------------------------------------------------------------------------
// Launch
// ---------------------------------------------------------------------------
extern "C" void kernel_launch(void* const* d_in, const int* in_sizes, int n_in,
                              void* d_out, int out_size, void* d_ws,
                              size_t ws_size, hipStream_t stream) {
  const float* x_user = (const float*)d_in[0];
  const float* x_item = (const float*)d_in[1];
  const float* Wv = (const float*)d_in[2];
  const float* bv = (const float*)d_in[3];
  const float* Wq = (const float*)d_in[4];
  const float* bq = (const float*)d_in[5];
  const float* Wk = (const float*)d_in[6];
  const float* bk = (const float*)d_in[7];
  const int* src1 = (const int*)d_in[8];
  const int* dst1 = (const int*)d_in[9];
  const int* src2 = (const int*)d_in[10];
  const int* dst2 = (const int*)d_in[11];
  const int* src3 = (const int*)d_in[12];
  const int* dst3 = (const int*)d_in[13];
  float* out = (float*)d_out;

  const int NU = in_sizes[0] / DIN;
  const int NI = in_sizes[1] / DIN;
  const int E = in_sizes[8];
  const int NMAX = (NU > NI) ? NU : NI;

  // ---- workspace carve-out ----
  size_t off = 0;
  auto carve = [&](size_t bytes) {
    void* p = (char*)d_ws + off;
    off += (bytes + 255) & ~(size_t)255;
    return p;
  };
  unsigned short* xu_bf = (unsigned short*)carve((size_t)NU * DIN * 2);
  unsigned short* xi_bf = (unsigned short*)carve((size_t)NI * DIN * 2);
  unsigned short* WvT = (unsigned short*)carve((size_t)3 * HF * DIN * 2);
  float* Wqp = (float*)carve((size_t)3 * DIN * NHEAD * 4);
  float* bqp = (float*)carve(3 * NHEAD * 4);
  float* Wkp = (float*)carve((size_t)3 * DIN * NHEAD * 4);
  float* bkp = (float*)carve(3 * NHEAD * 4);
  float* h0 = (float*)carve((size_t)NU * HF * 4);  // user src, rel0
  float* h1 = (float*)carve((size_t)NI * HF * 4);  // item src, rel1
  float* h2 = (float*)carve((size_t)NU * HF * 4);  // user src, rel2
  float* q0 = (float*)carve((size_t)NI * NHEAD * 4);
  float* k0 = (float*)carve((size_t)NU * NHEAD * 4);
  float* q1 = (float*)carve((size_t)NU * NHEAD * 4);
  float* k1 = (float*)carve((size_t)NI * NHEAD * 4);
  float* q2 = (float*)carve((size_t)NU * NHEAD * 4);
  float* k2 = (float*)carve((size_t)NU * NHEAD * 4);
  unsigned* mord = (unsigned*)carve((size_t)NMAX * NHEAD * 4);
  float* mf = (float*)carve((size_t)NMAX * NHEAD * 4);
  float* denom = (float*)carve((size_t)NMAX * NHEAD * 4);
  float* agg_user = (float*)carve((size_t)NU * HF * 4);
  float* agg_item = (float*)carve((size_t)NI * HF * 4);
  (void)ws_size;

  const int B = 256;
  auto blocks = [&](long n) { return (unsigned)((n + B - 1) / B); };
  const unsigned ORD_NEG_INF = 0x007FFFFFu;  // f2ord(-inf)

  // ---- prep ----
  cvt_bf16_kernel<<<blocks((long)NU * DIN), B, 0, stream>>>(x_user, xu_bf, NU * DIN);
  cvt_bf16_kernel<<<blocks((long)NI * DIN), B, 0, stream>>>(x_item, xi_bf, NI * DIN);
  transpose_w_kernel<<<blocks(3L * HF * DIN), B, 0, stream>>>(Wv, WvT);
  fuse_w_kernel<<<blocks(3L * DIN * NHEAD), B, 0, stream>>>(Wv, Wq, Wqp);
  fuse_w_kernel<<<blocks(3L * DIN * NHEAD), B, 0, stream>>>(Wv, Wk, Wkp);
  fuse_b_kernel<<<1, 32, 0, stream>>>(bv, Wq, bq, bqp);
  fuse_b_kernel<<<1, 32, 0, stream>>>(bv, Wk, bk, bkp);

  // ---- big GEMMs via WMMA (src-side h only; block = 16 rows x full width) --
  gemm_bf16_wmma_kernel<<<NU / 16, B, 0, stream>>>(xu_bf, WvT + 0 * HF * DIN, bv + 0 * HF, h0, NU);
  gemm_bf16_wmma_kernel<<<NI / 16, B, 0, stream>>>(xi_bf, WvT + 1 * HF * DIN, bv + 1 * HF, h1, NI);
  gemm_bf16_wmma_kernel<<<NU / 16, B, 0, stream>>>(xu_bf, WvT + 2 * HF * DIN, bv + 2 * HF, h2, NU);

  // ---- fused skinny q/k GEMMs ----
  qk_gemm_kernel<<<blocks((long)NI * NHEAD), B, 0, stream>>>(x_item, Wqp + 0 * DIN * NHEAD, bqp + 0, q0, NI);
  qk_gemm_kernel<<<blocks((long)NU * NHEAD), B, 0, stream>>>(x_user, Wkp + 0 * DIN * NHEAD, bkp + 0, k0, NU);
  qk_gemm_kernel<<<blocks((long)NU * NHEAD), B, 0, stream>>>(x_user, Wqp + 1 * DIN * NHEAD, bqp + 8, q1, NU);
  qk_gemm_kernel<<<blocks((long)NI * NHEAD), B, 0, stream>>>(x_item, Wkp + 1 * DIN * NHEAD, bkp + 8, k1, NI);
  qk_gemm_kernel<<<blocks((long)NU * NHEAD), B, 0, stream>>>(x_user, Wqp + 2 * DIN * NHEAD, bqp + 16, q2, NU);
  qk_gemm_kernel<<<blocks((long)NU * NHEAD), B, 0, stream>>>(x_user, Wkp + 2 * DIN * NHEAD, bkp + 16, k2, NU);

  // ---- zero accumulators ----
  init_f32_kernel<<<blocks((long)NU * HF), B, 0, stream>>>(agg_user, 0.f, NU * HF);
  init_f32_kernel<<<blocks((long)NI * HF), B, 0, stream>>>(agg_item, 0.f, NI * HF);

  // ---- per-relation softmax + scatter (sequential; m/denom reused) ----
  struct Rel {
    const int* src; const int* dst; const float* q; const float* k;
    const float* h; float* agg; int nd;
  } rels[3] = {
      {src1, dst1, q0, k0, h0, agg_item, NI},
      {src2, dst2, q1, k1, h1, agg_user, NU},
      {src3, dst3, q2, k2, h2, agg_user, NU},
  };
  for (int r = 0; r < 3; ++r) {
    const Rel& rl = rels[r];
    int nseg = rl.nd * NHEAD;
    init_u32_kernel<<<blocks(nseg), B, 0, stream>>>(mord, ORD_NEG_INF, nseg);
    edge_max_kernel<<<blocks((long)E * NHEAD), B, 0, stream>>>(rl.src, rl.dst, rl.q, rl.k, mord, E);
    m_final_kernel<<<blocks(nseg), B, 0, stream>>>(mord, mf, denom, nseg);
    edge_sum_kernel<<<blocks((long)E * NHEAD), B, 0, stream>>>(rl.src, rl.dst, rl.q, rl.k, mf, denom, E);
    edge_scatter_kernel<<<(E + 7) / 8, B, 0, stream>>>(rl.src, rl.dst, rl.q, rl.k, mf, denom, rl.h, rl.agg, E);
  }

  // ---- finalize: out_user = mean_h(agg_user)/2 ; out_item = mean_h(agg_item)
  head_mean_kernel<<<blocks((long)NU * FOUT), B, 0, stream>>>(agg_user, out, 1.0f / 16.0f, NU);
  head_mean_kernel<<<blocks((long)NI * FOUT), B, 0, stream>>>(agg_item, out + (size_t)NU * FOUT, 1.0f / 8.0f, NI);
}
